// omninet_early_stop_14663018348604
// MI455X (gfx1250) — matrix operand order repaired and checked by reading, not compiled, not run on hardware
//
#include <hip/hip_runtime.h>
#include <hip/hip_bf16.h>
#include <stdint.h>

// ---------------- model dims ----------------
constexpr int TT    = 43;    // time steps
constexpr int NPIX  = 49;    // pixels
constexpr int DM    = 300;   // model dim
constexpr int H     = 8;     // heads
constexpr int HDK   = 512;   // H*DK
constexpr int DFC   = 100;   // mlp hidden
constexpr int DIMG  = 512;
constexpr int DTXT  = 300;
constexpr int BB    = 1024;  // batch
constexpr int LDT   = 320;   // padded DM stride (mult of 32)
constexpr int NW    = 8;     // waves per block (256 threads, wave32)

typedef __attribute__((ext_vector_type(16))) _Float16 v16h;
typedef __attribute__((ext_vector_type(8)))  float    v8f;

struct Params {
  const float *images, *captions;
  const float *W_img, *b_img, *W_cap, *b_cap;
  const float *Wq_t, *bq_t, *Wk_t, *bk_t, *Wv_t, *bv_t, *Wo_t, *bo_t, *g_t, *beta_t;
  const float *Wq_g, *bq_g, *Wk_g, *bk_g, *Wv_g, *bv_g, *Wo_g, *bo_g, *g_g, *beta_g;
  const float *Wc1, *bc1, *Wc2, *bc2, *Wa1, *ba1, *Wa2, *ba2;
  float* out;
};

__device__ __forceinline__ int laneid() { return (int)(threadIdx.x & 31u); }

__device__ __forceinline__ v8f wmma_f16(v16h a, v16h b, v8f c) {
  return __builtin_amdgcn_wmma_f32_16x16x32_f16(false, a, false, b, (short)0, c, false, false);
}

// =================== LDS staging (coalesced global -> LDS f16) ===================

// Stage W[k][n] (global f32, ld=ldw) transposed into dstT[(n-n0)*KB + (k-k0)] f16.
// Zero-fills k>=Kreal or n>=Nreal. Each thread reads 4 consecutive n (float4).
__device__ __forceinline__ void stage_wT(const float* __restrict__ W, int ldw,
                                         int Kreal, int Nreal,
                                         int k0, int KB, int n0, int NB,
                                         _Float16* dstT) {
  const int tid = threadIdx.x;
  const int nq = NB >> 2;
  for (int i = tid; i < KB * nq; i += 256) {
    int kk = i / nq;
    int nn = (i - kk * nq) << 2;
    int k  = k0 + kk;
    int n  = n0 + nn;
    float v0, v1, v2, v3;
    if (k < Kreal && (n + 3) < Nreal) {
      const float4 f = *reinterpret_cast<const float4*>(W + (size_t)k * ldw + n);
      v0 = f.x; v1 = f.y; v2 = f.z; v3 = f.w;
    } else {
      v0 = (k < Kreal && n     < Nreal) ? W[(size_t)k * ldw + n]     : 0.f;
      v1 = (k < Kreal && n + 1 < Nreal) ? W[(size_t)k * ldw + n + 1] : 0.f;
      v2 = (k < Kreal && n + 2 < Nreal) ? W[(size_t)k * ldw + n + 2] : 0.f;
      v3 = (k < Kreal && n + 3 < Nreal) ? W[(size_t)k * ldw + n + 3] : 0.f;
    }
    dstT[(nn    ) * KB + kk] = (_Float16)v0;
    dstT[(nn + 1) * KB + kk] = (_Float16)v1;
    dstT[(nn + 2) * KB + kk] = (_Float16)v2;
    dstT[(nn + 3) * KB + kk] = (_Float16)v3;
  }
}

// Stage A[m][k] (global f32) row-major into dst[m*ldd + k] f16, zero padded.
__device__ __forceinline__ void stage_a(const float* __restrict__ A, int lda,
                                        int Mreal, int Kreal,
                                        _Float16* dst, int ldd, int Mp) {
  const int tid = threadIdx.x;
  const int kq = ldd >> 2;
  for (int i = tid; i < Mp * kq; i += 256) {
    int m  = i / kq;
    int kk = (i - m * kq) << 2;
    _Float16 h0 = (_Float16)0.f, h1 = h0, h2 = h0, h3 = h0;
    if (m < Mreal) {
      if (kk + 3 < Kreal) {
        const float4 f = *reinterpret_cast<const float4*>(A + (size_t)m * lda + kk);
        h0 = (_Float16)f.x; h1 = (_Float16)f.y; h2 = (_Float16)f.z; h3 = (_Float16)f.w;
      } else {
        if (kk     < Kreal) h0 = (_Float16)A[(size_t)m * lda + kk];
        if (kk + 1 < Kreal) h1 = (_Float16)A[(size_t)m * lda + kk + 1];
        if (kk + 2 < Kreal) h2 = (_Float16)A[(size_t)m * lda + kk + 2];
        if (kk + 3 < Kreal) h3 = (_Float16)A[(size_t)m * lda + kk + 3];
      }
    }
    _Float16* q = dst + m * ldd + kk;
    q[0] = h0; q[1] = h1; q[2] = h2; q[3] = h3;
  }
}

// Stage a f32 vector into LDS f32, zero padded.
__device__ __forceinline__ void stage_vec(const float* __restrict__ src, int n,
                                          float* dst, int npad) {
  for (int i = threadIdx.x; i < npad; i += 256) dst[i] = (i < n) ? src[i] : 0.f;
}

// =================== WMMA fragment loaders (LDS only, contiguous) ===================

// A fragment 16x32 f16, row-major LDS (2 x 16B contiguous per lane)
__device__ __forceinline__ v16h a_frag_lh(const _Float16* A, int lda, int m0, int k0) {
  int lane = laneid();
  const _Float16* row = A + (m0 + (lane & 15)) * lda + k0 + ((lane & 16) ? 8 : 0);
  v16h a;
#pragma unroll
  for (int e = 0; e < 8; ++e) { a[e] = row[e]; a[8 + e] = row[16 + e]; }
  return a;
}

// A fragment from f32 LDS (attention probabilities)
__device__ __forceinline__ v16h a_frag_lf32(const float* A, int lda, int m0, int k0) {
  int lane = laneid();
  const float* row = A + (m0 + (lane & 15)) * lda + k0 + ((lane & 16) ? 8 : 0);
  v16h a;
#pragma unroll
  for (int e = 0; e < 8; ++e) {
    a[e]     = (_Float16)row[e];
    a[8 + e] = (_Float16)row[16 + e];
  }
  return a;
}

// B fragment 32x16 where B[k][n] = M[n][k] (M row-major [N][K] in LDS).
// Per-lane: 2 x 8 contiguous halfs (k = 8h..8h+7 and 16+8h..16+8h+7).
__device__ __forceinline__ v16h bt_frag_lh(const _Float16* Mm, int ldm, int k0, int n0) {
  int lane = laneid();
  const _Float16* row = Mm + (n0 + (lane & 15)) * ldm + k0 + ((lane & 16) ? 8 : 0);
  v16h b;
#pragma unroll
  for (int e = 0; e < 8; ++e) { b[e] = row[e]; b[8 + e] = row[16 + e]; }
  return b;
}

// =================== D-fragment stores ===================

// plain + bias (bias in LDS f32, indexed by local n), full 16x16
__device__ __forceinline__ void store_tile_bias(v8f d, _Float16* out, int ldo,
                                                int m0, int n0, const float* bias) {
  int lane = laneid();
  int n  = n0 + (lane & 15);
  int mo = (lane & 16) ? 8 : 0;
  float bv = bias[n];
#pragma unroll
  for (int g = 0; g < 8; ++g) out[(m0 + mo + g) * ldo + n] = (_Float16)(d[g] + bv);
}

// transposed store: out[n][m] (for V^T)
__device__ __forceinline__ void store_tile_biasT(v8f d, _Float16* out, int ldo,
                                                 int m0, int n0, const float* bias) {
  int lane = laneid();
  int n  = n0 + (lane & 15);
  int mo = (lane & 16) ? 8 : 0;
  float bv = bias[n];
#pragma unroll
  for (int g = 0; g < 8; ++g) out[n * ldo + (m0 + mo + g)] = (_Float16)(d[g] + bv);
}

// bounded store with bias and optional residual (resid == out), pad -> 0
template <bool RESID>
__device__ __forceinline__ void store_tile_bnd(v8f d, _Float16* out, int ldo,
                                               int m0, int n0, const float* bias,
                                               int Mmax, int Nmax) {
  int lane = laneid();
  int n  = n0 + (lane & 15);
  int mo = (lane & 16) ? 8 : 0;
  float bv = bias[n];
#pragma unroll
  for (int g = 0; g < 8; ++g) {
    int m = m0 + mo + g;
    if (m >= 64) continue;
    float v = 0.f;
    if (m < Mmax && n < Nmax) {
      v = d[g] + bv;
      if constexpr (RESID) v += (float)out[m * ldo + n];
    }
    out[m * ldo + n] = (_Float16)v;
  }
}

// scores store: f32 LDS, scale, optional per-row gate
template <bool USE_GATE>
__device__ __forceinline__ void store_tile_sc(v8f d, float* out, int ldo,
                                              int m0, int n0, float mul,
                                              const float* rowmul) {
  int lane = laneid();
  int n  = n0 + (lane & 15);
  int mo = (lane & 16) ? 8 : 0;
#pragma unroll
  for (int g = 0; g < 8; ++g) {
    int m = m0 + mo + g;
    float v = d[g] * mul;
    if constexpr (USE_GATE) v *= rowmul[m];
    out[m * ldo + n] = v;
  }
}

// relu(d + bias) -> f32 LDS, pad -> 0
__device__ __forceinline__ void store_tile_relu(v8f d, float* out, int ldo,
                                                int m0, int n0, const float* bias,
                                                int Mmax, int Nmax) {
  int lane = laneid();
  int n  = n0 + (lane & 15);
  int mo = (lane & 16) ? 8 : 0;
  float bv = bias[n];
#pragma unroll
  for (int g = 0; g < 8; ++g) {
    int m = m0 + mo + g;
    float v = 0.f;
    if (m < Mmax && n < Nmax) v = fmaxf(d[g] + bv, 0.f);
    out[m * ldo + n] = v;
  }
}

// =================== GEMM building blocks ===================

// 64x64 = A(64x320, LDS f16) * W^T-slab(64x320) + bias; TR: store transposed
template <bool TR>
__device__ __forceinline__ void qkv_gemm(const _Float16* Asrc, const _Float16* wT,
                                         const float* bias, _Float16* dst) {
  const int wave = (int)(threadIdx.x >> 5);
#pragma unroll
  for (int t = 0; t < 2; ++t) {
    int tile = wave + t * NW;
    int m0 = (tile >> 2) * 16, n0 = (tile & 3) * 16;
    v8f acc = {};
#pragma unroll
    for (int ks = 0; ks < 10; ++ks) {
      v16h a = a_frag_lh(Asrc, LDT, m0, ks * 32);
      v16h b = bt_frag_lh(wT, 320, ks * 32, n0);
      acc = wmma_f16(a, b, acc);
    }
    if constexpr (TR) store_tile_biasT(acc, dst, 64, m0, n0, bias);
    else              store_tile_bias (acc, dst, 64, m0, n0, bias);
  }
}

// =================== attention layer ===================

template <bool CAUSAL, bool SAVE_GATE, bool USE_GATE>
__device__ void attn_layer(_Float16* tc, const _Float16* kvsrc,
                           _Float16* qh, _Float16* kh, _Float16* vhT,
                           float* sc, float* gate, _Float16* ao,
                           _Float16* wbuf, float* fbuf,
                           const float* Wq, const float* bq,
                           const float* Wk, const float* bk,
                           const float* Wv, const float* bv,
                           const float* Wo, const float* bo,
                           const float* gvec, const float* bvec,
                           int kcols) {
  const int tid = threadIdx.x, wave = tid >> 5, lane = tid & 31;
  const float scale = 0.125f; // 1/sqrt(64)

  for (int h = 0; h < H; ++h) {
    // ---- Q ----
    stage_wT(Wq, HDK, DM, HDK, 0, 320, h * 64, 64, wbuf);
    stage_vec(bq + h * 64, 64, fbuf, 64);
    __syncthreads();
    qkv_gemm<false>(tc, wbuf, fbuf, qh);
    __syncthreads();
    // ---- K ----
    stage_wT(Wk, HDK, DM, HDK, 0, 320, h * 64, 64, wbuf);
    stage_vec(bk + h * 64, 64, fbuf, 64);
    __syncthreads();
    qkv_gemm<false>(kvsrc, wbuf, fbuf, kh);
    __syncthreads();
    // ---- V (transposed store) ----
    stage_wT(Wv, HDK, DM, HDK, 0, 320, h * 64, 64, wbuf);
    stage_vec(bv + h * 64, 64, fbuf, 64);
    __syncthreads();
    qkv_gemm<true>(kvsrc, wbuf, fbuf, vhT);
    __syncthreads();

    // ---- scores = Q * K^T * scale (* gate) ----
#pragma unroll
    for (int t = 0; t < 2; ++t) {
      int tile = wave + t * NW;
      int m0 = (tile >> 2) * 16, n0 = (tile & 3) * 16;
      v8f acc = {};
#pragma unroll
      for (int ks = 0; ks < 2; ++ks) {
        v16h a = a_frag_lh(qh, 64, m0, ks * 32);
        v16h b = bt_frag_lh(kh, 64, ks * 32, n0);
        acc = wmma_f16(a, b, acc);
      }
      store_tile_sc<USE_GATE>(acc, sc, 64, m0, n0, scale, gate + h * 64);
    }
    __syncthreads();

    // ---- softmax (row per thread), causal, gate capture, zero padding ----
    if (tid < 64) {
      int q = tid;
      float* row = sc + q * 64;
      if (q < TT) {
        int kmax = CAUSAL ? (q + 1) : kcols;
        float mx = -1e30f;
        for (int k2 = 0; k2 < kmax; ++k2) mx = fmaxf(mx, row[k2]);
        float s = 0.f;
        for (int k2 = 0; k2 < kmax; ++k2) { float e = __expf(row[k2] - mx); row[k2] = e; s += e; }
        float inv = 1.f / s;
        for (int k2 = 0; k2 < kmax; ++k2) row[k2] *= inv;
        for (int k2 = kmax; k2 < 64; ++k2) row[k2] = 0.f;
        if constexpr (SAVE_GATE) gate[h * 64 + q] = row[0];
      } else {
        for (int k2 = 0; k2 < 64; ++k2) row[k2] = 0.f;
      }
    }
    __syncthreads();

    // ---- out_h = attn * V -> ao[:, h*64 .. ] ----
#pragma unroll
    for (int t = 0; t < 2; ++t) {
      int tile = wave + t * NW;
      int m0 = (tile >> 2) * 16, n0 = (tile & 3) * 16;
      v8f acc = {};
#pragma unroll
      for (int ks = 0; ks < 2; ++ks) {
        v16h a = a_frag_lf32(sc, 64, m0, ks * 32);
        v16h b = bt_frag_lh(vhT, 64, ks * 32, n0);
        acc = wmma_f16(a, b, acc);
      }
      // store without bias: reuse biased store with zero via fbuf? use plain loop
      {
        int ln = laneid();
        int n = h * 64 + n0 + (ln & 15);
        int mo = (ln & 16) ? 8 : 0;
#pragma unroll
        for (int g = 0; g < 8; ++g) ao[(m0 + mo + g) * 512 + n] = (_Float16)acc[g];
      }
    }
    __syncthreads();
  }

  // ---- out = ao @ Wo + bo + residual(tc) -> tc ----
  stage_vec(bo, DM, fbuf, LDT);
  v8f acc[10];
  v8f zero = {};
#pragma unroll
  for (int t = 0; t < 10; ++t) acc[t] = zero;
  for (int kb = 0; kb < 4; ++kb) {
    stage_wT(Wo, DM, HDK, DM, kb * 128, 128, 0, LDT, wbuf);
    __syncthreads();
#pragma unroll
    for (int t = 0; t < 10; ++t) {
      int tile = wave + t * NW;
      int m0 = (tile / 20) * 16, n0 = (tile % 20) * 16;
#pragma unroll
      for (int ks = 0; ks < 4; ++ks) {
        v16h a = a_frag_lh(ao, 512, m0, kb * 128 + ks * 32);
        v16h b = bt_frag_lh(wbuf, 128, ks * 32, n0);
        acc[t] = wmma_f16(a, b, acc[t]);
      }
    }
    __syncthreads();
  }
#pragma unroll
  for (int t = 0; t < 10; ++t) {
    int tile = wave + t * NW;
    store_tile_bnd<true>(acc[t], tc, LDT, (tile / 20) * 16, (tile % 20) * 16,
                         fbuf, TT, DM);
  }
  __syncthreads();

  // ---- LayerNorm (wave per row, lane-parallel + shuffle reduction) ----
  stage_vec(gvec, DM, fbuf, LDT);
  stage_vec(bvec, DM, fbuf + LDT, LDT);
  __syncthreads();
  for (int r = wave; r < TT; r += NW) {
    float s = 0.f, s2 = 0.f;
    for (int d = lane; d < DM; d += 32) {
      float x = (float)tc[r * LDT + d];
      s += x; s2 += x * x;
    }
#pragma unroll
    for (int o = 16; o; o >>= 1) { s += __shfl_xor(s, o, 32); s2 += __shfl_xor(s2, o, 32); }
    float mean = s / DM;
    float var  = s2 / DM - mean * mean;
    float inv  = rsqrtf(var + 1e-5f);
    for (int d = lane; d < DM; d += 32) {
      float x = (float)tc[r * LDT + d];
      tc[r * LDT + d] = (_Float16)(((x - mean) * inv) * fbuf[d] + fbuf[LDT + d]);
    }
  }
  __syncthreads();
}

// =================== classifier head ===================

__device__ void head_gemm(const _Float16* tc, const float* W1, const float* b1,
                          const float* W2, const float* b2,
                          _Float16* wbuf, float* fbuf, float* hb,
                          float* outp, int b) {
  const int tid = threadIdx.x, wave = tid >> 5;
  // hidden = relu(tc @ W1 + b1): full W1^T slab (112 x 320 f16)
  stage_wT(W1, DFC, DM, DFC, 0, 320, 0, 112, wbuf);
  stage_vec(b1, DFC, fbuf, 112);
  __syncthreads();
#pragma unroll
  for (int t = 0; t < 4; ++t) {
    int tile = wave + t * NW;
    if (tile < 28) {
      int m0 = (tile / 7) * 16, n0 = (tile % 7) * 16;
      v8f acc = {};
#pragma unroll
      for (int ks = 0; ks < 10; ++ks) {
        v16h a  = a_frag_lh(tc, LDT, m0, ks * 32);
        v16h bf = bt_frag_lh(wbuf, 320, ks * 32, n0);
        acc = wmma_f16(a, bf, acc);
      }
      store_tile_relu(acc, hb, 112, m0, n0, fbuf, TT, DFC);
    }
  }
  __syncthreads();
  // logits = hidden @ W2 + b2 (N=2: VALU dot)
  for (int i = tid; i < DFC * 2; i += 256) fbuf[i] = W2[i];
  if (tid < 2) fbuf[DFC * 2 + tid] = b2[tid];
  __syncthreads();
  for (int idx = tid; idx < TT * 2; idx += 256) {
    int t = idx >> 1, c = idx & 1;
    float acc = fbuf[DFC * 2 + c];
    for (int j = 0; j < DFC; ++j) acc += hb[t * 112 + j] * fbuf[j * 2 + c];
    outp[((size_t)b * TT + t) * 2 + c] = acc;
  }
  __syncthreads();
}

// =================== fused kernel ===================

__global__ __launch_bounds__(256)
void omninet_wmma_fused_kernel(Params p) {
  extern __shared__ char smem[];
  _Float16* tc   = (_Float16*)smem;         // 64x320 f16  residual stream      (40960 B)
  _Float16* sp   = tc + 64 * LDT;           // 64x320 f16  spatial cache        (40960 B)
  _Float16* ao   = sp + 64 * LDT;           // 64x512 f16  attn out / staged img(65536 B)
  _Float16* qh   = ao + 64 * 512;           // 64x64 f16   Q                    ( 8192 B)
  _Float16* kh   = qh + 64 * 64;            // 64x64 f16   K                    ( 8192 B)
  _Float16* vhT  = kh + 64 * 64;            // 64x64 f16   V transposed         ( 8192 B)
  float*    sc   = (float*)(vhT + 64 * 64); // 64x64 f32   scores/probs         (16384 B)
  float*    gate = sc + 64 * 64;            // 8x64  f32   gate                 ( 2048 B)
  float*    hb   = gate + 8 * 64;           // 64x112 f32  mlp hidden           (28672 B)
  float*    fbuf = hb + 64 * 112;           // 1024  f32   bias/LN/head params  ( 4096 B)
  _Float16* wbuf = (_Float16*)(fbuf + 1024);// 40960 f16   weight slab          (81920 B)
  _Float16* cbuf = qh;                      // overlay: staged captions 64x320 (qh..sc)

  const int tid = threadIdx.x, wave = tid >> 5;
  const int b = blockIdx.x;

  // L2 warm for shared weights (global_prefetch_b8)
  if (tid < 32) {
    __builtin_prefetch(p.Wq_t + tid * 4096, 0, 1);
    __builtin_prefetch(p.Wo_t + tid * 4096, 0, 1);
    __builtin_prefetch(p.Wq_g + tid * 4096, 0, 1);
    __builtin_prefetch(p.Wo_g + tid * 4096, 0, 1);
    __builtin_prefetch(p.W_img + tid * 4096, 0, 1);
  }

  // stage activations; zero tc + gate
  stage_a(p.images + (size_t)b * NPIX * DIMG, DIMG, NPIX, DIMG, ao, 512, 64);
  stage_a(p.captions + (size_t)b * (TT - 1) * DTXT, DTXT, TT - 1, DTXT, cbuf, LDT, 64);
  {
    uint32_t* z = (uint32_t*)tc;
    for (int i = tid; i < 64 * LDT / 2; i += 256) z[i] = 0u;
    for (int i = tid; i < 8 * 64; i += 256) gate[i] = 0.f;
  }
  stage_vec(p.b_img, DM, fbuf, LDT);
  __syncthreads();

  // ---- P1: spatial = images @ W_img + b_img (A = staged images in ao) ----
  {
    v8f acc[10];
    v8f zero = {};
#pragma unroll
    for (int t = 0; t < 10; ++t) acc[t] = zero;
    for (int kb = 0; kb < 4; ++kb) {
      stage_wT(p.W_img, DM, DIMG, DM, kb * 128, 128, 0, LDT, wbuf);
      __syncthreads();
#pragma unroll
      for (int t = 0; t < 10; ++t) {
        int tile = wave + t * NW;
        int m0 = (tile / 20) * 16, n0 = (tile % 20) * 16;
#pragma unroll
        for (int ks = 0; ks < 4; ++ks) {
          v16h a = a_frag_lh(ao, 512, m0, kb * 128 + ks * 32);
          v16h bf = bt_frag_lh(wbuf, 128, ks * 32, n0);
          acc[t] = wmma_f16(a, bf, acc[t]);
        }
      }
      __syncthreads();
    }
#pragma unroll
    for (int t = 0; t < 10; ++t) {
      int tile = wave + t * NW;
      store_tile_bnd<false>(acc[t], sp, LDT, (tile / 20) * 16, (tile % 20) * 16,
                            fbuf, NPIX, DM);
    }
  }
  __syncthreads();

  // ---- P2: tc[1..42] = captions @ W_cap + b_cap (A = staged captions) ----
  {
    stage_vec(p.b_cap, DM, fbuf, LDT);
    v8f acc[8];
    v8f zero = {};
#pragma unroll
    for (int t = 0; t < 8; ++t) acc[t] = zero;
    const int kb0s[3] = {0, 128, 256};
    const int kbsz[3] = {128, 128, 64};
    for (int kb = 0; kb < 3; ++kb) {
      __syncthreads();
      stage_wT(p.W_cap, DM, DTXT, DM, kb0s[kb], kbsz[kb], 0, LDT, wbuf);
      __syncthreads();
#pragma unroll
      for (int t = 0; t < 8; ++t) {
        int tile = wave + t * NW;
        if (tile < 60) {
          int m0 = (tile / 20) * 16, n0 = (tile % 20) * 16;
          for (int ks = 0; ks < kbsz[kb] / 32; ++ks) {
            v16h a = a_frag_lh(cbuf, LDT, m0, kb0s[kb] + ks * 32);
            v16h bf = bt_frag_lh(wbuf, kbsz[kb], ks * 32, n0);
            acc[t] = wmma_f16(a, bf, acc[t]);
          }
        }
      }
    }
    __syncthreads();
#pragma unroll
    for (int t = 0; t < 8; ++t) {
      int tile = wave + t * NW;
      if (tile < 60)
        store_tile_bnd<false>(acc[t], tc, LDT, (tile / 20) * 16 + 1, (tile % 20) * 16,
                              fbuf, TT, DM);
    }
  }
  __syncthreads();

  // ---- P3a: tc[0] = mean over pixels of spatial ----
  for (int d = tid; d < DM; d += 256) {
    float s = 0.f;
    for (int r = 0; r < NPIX; ++r) s += (float)sp[r * LDT + d];
    tc[d] = (_Float16)(s * (1.f / (float)NPIX));
  }
  __syncthreads();

  // ---- P3b: tc += sinusoid positional table ----
  for (int idx = tid; idx < TT * DM; idx += 256) {
    int r = idx / DM, d = idx % DM;
    float ang = (float)r * __powf(10000.f, -2.f * (float)(d / 2) / (float)DM);
    float pe = (d & 1) ? __cosf(ang) : __sinf(ang);
    tc[r * LDT + d] = (_Float16)((float)tc[r * LDT + d] + pe);
  }
  __syncthreads();

  // ---- P4: temporal causal self-attention + LN ----
  attn_layer<true, true, false>(tc, tc, qh, kh, vhT, sc, gate, ao, wbuf, fbuf,
                                p.Wq_t, p.bq_t, p.Wk_t, p.bk_t, p.Wv_t, p.bv_t,
                                p.Wo_t, p.bo_t, p.g_t, p.beta_t, TT);

  // ---- P5: gated cross-attention onto spatial + LN ----
  attn_layer<false, false, true>(tc, sp, qh, kh, vhT, sc, gate, ao, wbuf, fbuf,
                                 p.Wq_g, p.bq_g, p.Wk_g, p.bk_g, p.Wv_g, p.bv_g,
                                 p.Wo_g, p.bo_g, p.g_g, p.beta_g, NPIX);

  // ---- P6: classifier heads ----
  head_gemm(tc, p.Wc1, p.bc1, p.Wc2, p.bc2, wbuf, fbuf, hb, p.out, b);
  head_gemm(tc, p.Wa1, p.ba1, p.Wa2, p.ba2, wbuf, fbuf, hb,
            p.out + (size_t)BB * TT * 2, b);
}

extern "C" void kernel_launch(void* const* d_in, const int* in_sizes, int n_in,
                              void* d_out, int out_size, void* d_ws, size_t ws_size,
                              hipStream_t stream) {
  (void)in_sizes; (void)n_in; (void)d_ws; (void)ws_size; (void)out_size;
  Params p;
  p.images   = (const float*)d_in[0];
  p.captions = (const float*)d_in[1];
  p.W_img = (const float*)d_in[2];  p.b_img = (const float*)d_in[3];
  p.W_cap = (const float*)d_in[4];  p.b_cap = (const float*)d_in[5];
  p.Wq_t = (const float*)d_in[6];   p.bq_t = (const float*)d_in[7];
  p.Wk_t = (const float*)d_in[8];   p.bk_t = (const float*)d_in[9];
  p.Wv_t = (const float*)d_in[10];  p.bv_t = (const float*)d_in[11];
  p.Wo_t = (const float*)d_in[12];  p.bo_t = (const float*)d_in[13];
  p.g_t  = (const float*)d_in[14];  p.beta_t = (const float*)d_in[15];
  p.Wq_g = (const float*)d_in[16];  p.bq_g = (const float*)d_in[17];
  p.Wk_g = (const float*)d_in[18];  p.bk_g = (const float*)d_in[19];
  p.Wv_g = (const float*)d_in[20];  p.bv_g = (const float*)d_in[21];
  p.Wo_g = (const float*)d_in[22];  p.bo_g = (const float*)d_in[23];
  p.g_g  = (const float*)d_in[24];  p.beta_g = (const float*)d_in[25];
  p.Wc1 = (const float*)d_in[26];   p.bc1 = (const float*)d_in[27];
  p.Wc2 = (const float*)d_in[28];   p.bc2 = (const float*)d_in[29];
  p.Wa1 = (const float*)d_in[30];   p.ba1 = (const float*)d_in[31];
  p.Wa2 = (const float*)d_in[32];   p.ba2 = (const float*)d_in[33];
  p.out = (float*)d_out;

  // LDS: tc+sp+ao+qkv+sc+gate+hb+fbuf+wbuf = 305152 bytes (< 320KB WGP limit)
  const size_t smem_bytes =
      (size_t)(64 * LDT * 2 + 64 * 512 + 3 * 64 * 64) * sizeof(_Float16) +
      (size_t)(64 * 64 + 8 * 64 + 64 * 112 + 1024) * sizeof(float) +
      (size_t)40960 * sizeof(_Float16);

  omninet_wmma_fused_kernel<<<BB, 256, smem_bytes, stream>>>(p);
}
